// VarifoldLayer_28690381537760
// MI455X (gfx1250) — matrix operand order repaired and checked by reading, not compiled, not run on hardware
//
#include <hip/hip_runtime.h>
#include <hip/hip_bf16.h>

typedef _Float16 half_t;
typedef __attribute__((ext_vector_type(8)))  _Float16 v8h;
typedef __attribute__((ext_vector_type(16))) _Float16 v16h;
typedef __attribute__((ext_vector_type(2)))  float    v2f;
typedef __attribute__((ext_vector_type(8)))  float    v8f;

#define FEAT 128
#define NPTS 2048
#define NB   4

// ---------------------------------------------------------------------------
// WMMA helpers
// ---------------------------------------------------------------------------
__device__ __forceinline__ v8f wmma16(v16h a, v16h b, v8f c) {
  return __builtin_amdgcn_wmma_f32_16x16x32_f16(
      false, a, false, b, (short)0, c, false, false);
}

// D = A(16x4 f32) * B(4x16 f32) + C ; used for the position Gram matrix.
__device__ __forceinline__ v8f wmma4f32(v2f a, v2f b, v8f c) {
  return __builtin_amdgcn_wmma_f32_16x16x4_f32(
      false, a, false, b, (short)0, c, false, false);
}

// A fragment (16-bit A 16x32, ISA 7.12.2): lane m = lane&15, hl = lane>>4.
// halves 0..7  <- K = k0 + 8*hl + 0..7 ; halves 8..15 <- K = k0+16+8*hl+0..7
__device__ __forceinline__ v16h load_a(const half_t* base, int ld, int lane, int k0) {
  const int m = lane & 15, hl = lane >> 4;
  const v8h a0 = *(const v8h*)(base + m * ld + k0 + 8 * hl);
  const v8h a1 = *(const v8h*)(base + m * ld + k0 + 16 + 8 * hl);
  return __builtin_shufflevector(a0, a1, 0, 1, 2, 3, 4, 5, 6, 7,
                                 8, 9, 10, 11, 12, 13, 14, 15);
}

// B fragment (16-bit B 32x16): lane column n = n0 + (lane&15), halves 0..7
// <- K = k0 + 8*hl + 0..7 ; halves 8..15 <- K = k0+16+8*hl+0..7.  Source is K-major Bt[n][k].
__device__ __forceinline__ v16h load_b(const half_t* bt, int ldk, int lane,
                                       int k0, int n0) {
  const int n = lane & 15, hl = lane >> 4;
  const half_t* p = bt + (size_t)(n0 + n) * ldk + k0 + 8 * hl;
  const v8h b0 = *(const v8h*)p;
  const v8h b1 = *(const v8h*)(p + 16);
  return __builtin_shufflevector(b0, b1, 0, 1, 2, 3, 4, 5, 6, 7,
                                 8, 9, 10, 11, 12, 13, 14, 15);
}

// ---------------------------------------------------------------------------
// Prep kernels
// ---------------------------------------------------------------------------
__global__ void cvt_f16_kernel(const float* __restrict__ src,
                               half_t* __restrict__ dst, int n) {
  int i = blockIdx.x * 256 + threadIdx.x;
  if (i < n) dst[i] = (half_t)src[i];
}

// features (B,N,128) f32 -> Ft (B,128,N) f16 (K-major for aggregation B-frags)
__global__ void tfeat_kernel(const float* __restrict__ feats,
                             half_t* __restrict__ ft) {
  int i = blockIdx.x * 256 + threadIdx.x;
  if (i < NB * FEAT * NPTS) {
    int j = i & (NPTS - 1);
    int c = (i >> 11) & (FEAT - 1);
    int b = i >> 18;
    ft[i] = (half_t)feats[((size_t)(b * NPTS + j)) * FEAT + c];
  }
}

// W (K x Nn, row-major in k) f32 -> Wt (Nn x K) f16 (K-major)
__global__ void tw_kernel(const float* __restrict__ w, half_t* __restrict__ wt,
                          int K, int Nn) {
  int i = blockIdx.x * 256 + threadIdx.x;
  if (i < K * Nn) {
    int k = i % K;
    int n = i / K;
    wt[i] = (half_t)w[(size_t)k * Nn + n];
  }
}

// ---------------------------------------------------------------------------
// Kernel 1: fused Gaussian kernel + row-normalized aggregation.
// Block = 128 threads (4 waves) per (batch, 16-row tile).
// j-chunk = 128: each wave computes two 16x16 Gram tiles via
// v_wmma_f32_16x16x4_f32 (dist^2 = ri + rj - 2*G, exact f32), applies
// exp/weight pointwise into a shared 16x128 f16 kernel tile, then all waves
// consume it with v_wmma_f32_16x16x32_f16 against the transposed features.
// ---------------------------------------------------------------------------
__global__ void __launch_bounds__(128)
agg_kernel(const float* __restrict__ pos, const float* __restrict__ wts,
           const half_t* __restrict__ ft, half_t* __restrict__ agg16) {
  const int t = threadIdx.x;
  const int lane = t & 31, wv = t >> 5;
  const int b = blockIdx.x >> 7;          // 128 row-tiles per batch
  const int i0 = (blockIdx.x & 127) * 16;
  const int mloc = lane & 15, hl = lane >> 4;

  __shared__ __align__(16) half_t kbuf[16 * 128]; // f16 kernel tile (M x j)
  __shared__ float sri[16];    // |p_i|^2 for the row tile
  __shared__ float srow[16];   // kernel row sums
  __shared__ float srj[128];   // |p_j|^2 for current chunk
  __shared__ float swj[128];   // w_j for current chunk
  __shared__ float spx[128], spy[128], spz[128];

  if (t < 16) {
    const float* pi = pos + (size_t)(b * NPTS + i0 + t) * 3;
    sri[t] = pi[0] * pi[0] + pi[1] * pi[1] + pi[2] * pi[2];
    srow[t] = 0.0f;
  }
  __syncthreads();

  // Gram A fragment (loop-invariant): 32-bit A 16x4 layout.
  // lanes 0-15: K=0,1 (x,y); lanes 16-31: K=2,3 (z, pad0)
  v2f afrag;
  {
    const float* pi = pos + (size_t)(b * NPTS + i0 + mloc) * 3;
    afrag[0] = (hl == 0) ? pi[0] : pi[2];
    afrag[1] = (hl == 0) ? pi[1] : 0.0f;
  }
  float riv[8];
#pragma unroll
  for (int r = 0; r < 8; ++r) riv[r] = sri[r + 8 * hl];

  const half_t* ftb = ft + (size_t)b * FEAT * NPTS;
  const int cb = wv * 32;
  v8f acc0 = {};
  v8f acc1 = {};
  float rs[8];
#pragma unroll
  for (int r = 0; r < 8; ++r) rs[r] = 0.0f;

  for (int j0 = 0; j0 < NPTS; j0 += 128) {
    __syncthreads();  // previous chunk's reads of srj/kbuf are done
    {               // stage |p_j|^2, w_j, positions for j = j0 + t
      const float* pj = pos + (size_t)(b * NPTS + j0 + t) * 3;
      const float x = pj[0], y = pj[1], z = pj[2];
      spx[t] = x; spy[t] = y; spz[t] = z;
      srj[t] = x * x + y * y + z * z;
      swj[t] = wts[b * NPTS + j0 + t];
    }
    __syncthreads();

    // ---- Gram + pointwise exp into kbuf: wave wv owns j-tiles 2wv, 2wv+1 ---
#pragma unroll
    for (int tile = 0; tile < 2; ++tile) {
      const int t16 = (wv * 2 + tile) * 16;
      const int jl = t16 + mloc;           // local j column for this lane
      v2f bfrag;
      bfrag[0] = (hl == 0) ? spx[jl] : spz[jl];
      bfrag[1] = (hl == 0) ? spy[jl] : 0.0f;
      v8f g = {};
      g = wmma4f32(afrag, bfrag, g);       // G[M][N] = p_i . p_j (f32)
      const float rjv = srj[jl];
      const float wjv = swj[jl];
#pragma unroll
      for (int r = 0; r < 8; ++r) {
        const int M = r + 8 * hl;
        const float d2 = riv[r] + rjv - 2.0f * g[r];
        const float kv = __expf(-2.0f * d2) * wjv;  // exp(-d2/(2*0.25))
        rs[r] += kv;
        kbuf[M * 128 + jl] = (half_t)kv;
      }
    }
    __syncthreads();

    // ---- aggregation GEMM over this 128-wide K chunk ----
#pragma unroll
    for (int k0 = 0; k0 < 128; k0 += 32) {
      const v16h a = load_a(kbuf, 128, lane, k0);
      acc0 = wmma16(a, load_b(ftb, NPTS, lane, j0 + k0, cb), acc0);
      acc1 = wmma16(a, load_b(ftb, NPTS, lane, j0 + k0, cb + 16), acc1);
    }
  }

#pragma unroll
  for (int r = 0; r < 8; ++r) atomicAdd(&srow[r + 8 * hl], rs[r]);
  __syncthreads();

#pragma unroll
  for (int r = 0; r < 8; ++r) {
    const int M = r + 8 * hl;
    const float inv = 1.0f / (srow[M] + 1e-8f);
    const size_t gr = (size_t)(b * NPTS + i0 + M) * FEAT;
    agg16[gr + cb + mloc] = (half_t)(acc0[r] * inv);
    agg16[gr + cb + 16 + mloc] = (half_t)(acc1[r] * inv);
  }
}

// ---------------------------------------------------------------------------
// Kernel 2: fused MLP + gated attention + residual epilogue.
// ---------------------------------------------------------------------------
__global__ void __launch_bounds__(128)
mlp_kernel(const float* __restrict__ feats_f32, const half_t* __restrict__ feat16,
           const half_t* __restrict__ agg16,
           const half_t* __restrict__ w1t, const float* __restrict__ b1,
           const half_t* __restrict__ w2t, const float* __restrict__ b2,
           const half_t* __restrict__ wa1t, const float* __restrict__ ba1,
           const float* __restrict__ wa2, const float* __restrict__ ba2,
           float* __restrict__ out) {
  const int t = threadIdx.x;
  const int lane = t & 31, wv = t >> 5;
  const int m0 = blockIdx.x * 16;
  const int cb = wv * 32;
  const int mloc = lane & 15, hl = lane >> 4;
  const int n0 = cb + mloc, n1 = cb + 16 + mloc;

  __shared__ __align__(16) half_t hbuf[16 * FEAT];
  __shared__ __align__(16) half_t tbuf[16 * FEAT];
  __shared__ float attsum[16];
  __shared__ float attv[16];

  if (t < 16) attsum[t] = 0.0f;

  // ---- h = relu(agg @ W1 + b1) ----
  const half_t* arow = agg16 + (size_t)m0 * FEAT;
  v8f h0 = {}, h1 = {};
#pragma unroll
  for (int k0 = 0; k0 < FEAT; k0 += 32) {
    const v16h a = load_a(arow, FEAT, lane, k0);
    h0 = wmma16(a, load_b(w1t, FEAT, lane, k0, cb), h0);
    h1 = wmma16(a, load_b(w1t, FEAT, lane, k0, cb + 16), h1);
  }
  {
    const float bb0 = b1[n0], bb1 = b1[n1];
#pragma unroll
    for (int r = 0; r < 8; ++r) {
      const int M = r + 8 * hl;
      hbuf[M * FEAT + n0] = (half_t)fmaxf(h0[r] + bb0, 0.0f);
      hbuf[M * FEAT + n1] = (half_t)fmaxf(h1[r] + bb1, 0.0f);
    }
  }
  __syncthreads();

  // ---- transformed = h @ W2 + b2 ----
  v8f t0 = {}, t1 = {};
#pragma unroll
  for (int k0 = 0; k0 < FEAT; k0 += 32) {
    const v16h a = load_a(hbuf, FEAT, lane, k0);
    t0 = wmma16(a, load_b(w2t, FEAT, lane, k0, cb), t0);
    t1 = wmma16(a, load_b(w2t, FEAT, lane, k0, cb + 16), t1);
  }
  float tr0[8], tr1[8];
  {
    const float bb0 = b2[n0], bb1 = b2[n1];
#pragma unroll
    for (int r = 0; r < 8; ++r) {
      const int M = r + 8 * hl;
      tr0[r] = t0[r] + bb0;
      tr1[r] = t1[r] + bb1;
      tbuf[M * FEAT + n0] = (half_t)tr0[r];
      tbuf[M * FEAT + n1] = (half_t)tr1[r];
    }
  }
  __syncthreads();

  // ---- a = relu([features, transformed] @ Wa1 + ba1) ----  (K = 256)
  const half_t* frow = feat16 + (size_t)m0 * FEAT;
  v8f a0 = {}, a1 = {};
#pragma unroll
  for (int k0 = 0; k0 < FEAT; k0 += 32) {
    const v16h a = load_a(frow, FEAT, lane, k0);
    a0 = wmma16(a, load_b(wa1t, 2 * FEAT, lane, k0, cb), a0);
    a1 = wmma16(a, load_b(wa1t, 2 * FEAT, lane, k0, cb + 16), a1);
  }
#pragma unroll
  for (int k0 = 0; k0 < FEAT; k0 += 32) {
    const v16h a = load_a(tbuf, FEAT, lane, k0);
    a0 = wmma16(a, load_b(wa1t, 2 * FEAT, lane, FEAT + k0, cb), a0);
    a1 = wmma16(a, load_b(wa1t, 2 * FEAT, lane, FEAT + k0, cb + 16), a1);
  }

  // ---- attention = sigmoid(a @ Wa2 + ba2) ----
  {
    const float w20 = wa2[n0], w21 = wa2[n1];
    const float bv0 = ba1[n0], bv1 = ba1[n1];
#pragma unroll
    for (int r = 0; r < 8; ++r) {
      const int M = r + 8 * hl;
      const float p = fmaxf(a0[r] + bv0, 0.0f) * w20 +
                      fmaxf(a1[r] + bv1, 0.0f) * w21;
      atomicAdd(&attsum[M], p);
    }
  }
  __syncthreads();
  if (t < 16) attv[t] = 1.0f / (1.0f + __expf(-(attsum[t] + ba2[0])));
  __syncthreads();

  // ---- out = features + attention * transformed ----
#pragma unroll
  for (int r = 0; r < 8; ++r) {
    const int M = r + 8 * hl;
    const size_t gr = (size_t)(m0 + M) * FEAT;
    const float att = attv[M];
    out[gr + n0] = feats_f32[gr + n0] + att * tr0[r];
    out[gr + n1] = feats_f32[gr + n1] + att * tr1[r];
  }
}

// ---------------------------------------------------------------------------
extern "C" void kernel_launch(void* const* d_in, const int* in_sizes, int n_in,
                              void* d_out, int out_size, void* d_ws, size_t ws_size,
                              hipStream_t stream) {
  const float* positions = (const float*)d_in[0];
  const float* features  = (const float*)d_in[1];
  const float* weights   = (const float*)d_in[2];
  const float* W1  = (const float*)d_in[3];
  const float* b1  = (const float*)d_in[4];
  const float* W2  = (const float*)d_in[5];
  const float* b2  = (const float*)d_in[6];
  const float* Wa1 = (const float*)d_in[7];
  const float* ba1 = (const float*)d_in[8];
  const float* Wa2 = (const float*)d_in[9];
  const float* ba2 = (const float*)d_in[10];
  float* out = (float*)d_out;

  const size_t nfe = (size_t)NB * NPTS * FEAT;  // 1,048,576 elements
  half_t* ws     = (half_t*)d_ws;
  half_t* feat16 = ws;                 // row-major f16 features
  half_t* Ft     = feat16 + nfe;       // (B, 128, N) K-major f16 features
  half_t* agg16  = Ft + nfe;           // aggregated, row-major f16
  half_t* W1t    = agg16 + nfe;        // (out, in) f16
  half_t* W2t    = W1t + FEAT * FEAT;
  half_t* Wa1t   = W2t + FEAT * FEAT;  // (128, 256) f16

  cvt_f16_kernel<<<(int)((nfe + 255) / 256), 256, 0, stream>>>(features, feat16, (int)nfe);
  tfeat_kernel<<<(int)((nfe + 255) / 256), 256, 0, stream>>>(features, Ft);
  tw_kernel<<<(FEAT * FEAT + 255) / 256, 256, 0, stream>>>(W1, W1t, FEAT, FEAT);
  tw_kernel<<<(FEAT * FEAT + 255) / 256, 256, 0, stream>>>(W2, W2t, FEAT, FEAT);
  tw_kernel<<<(2 * FEAT * FEAT + 255) / 256, 256, 0, stream>>>(Wa1, Wa1t, 2 * FEAT, FEAT);

  agg_kernel<<<NB * (NPTS / 16), 128, 0, stream>>>(positions, weights, Ft, agg16);
  mlp_kernel<<<(NB * NPTS) / 16, 128, 0, stream>>>(features, feat16, agg16,
                                                   W1t, b1, W2t, b2,
                                                   Wa1t, ba1, Wa2, ba2, out);
}
